// WorldModelMamba_53558242181319
// MI455X (gfx1250) — matrix operand order repaired and checked by reading, not compiled, not run on hardware
//
#include <hip/hip_runtime.h>
#include <hip/hip_bf16.h>
#include <math.h>

// ---------------------------------------------------------------------------
// MI455X (gfx1250) world-model forward pass, round 5.
// * Conv encoder: implicit-GEMM WMMA bf16, NHWC-bf16 activations, tap-major
//   bf16 weights (zero-padded in K *and* output channels) + zero-padded bias
//   -> inner loop: 4x global_load_b128 + 1 wmma; epilogue fully branchless:
//   2x float4 bias loads + 8 gelu + 1 packed b128 store.
// * GEMMs: both operands pre-staged bf16; float4 bias/stores.
// * gelu = x*sigmoid(1.5958(x+0.044715x^3)) (exact tanh identity), sigmoid via
//   v_exp_f32 + v_rcp_f32 (__builtin_amdgcn_rcpf): branchless.
// * Batch-1 MLP: memory-bound (270MB weights / 23.3 TB/s) -> float4 GEMV.
// * Mamba2 scan in closed form: one 16x16 @ 16x64 WMMA per head.
// ---------------------------------------------------------------------------

typedef __attribute__((ext_vector_type(16))) __bf16 v16bf;
typedef __attribute__((ext_vector_type(8)))  float  v8f;
struct U4x2 { uint4 a, b; };

#define WPB 4   // waves per block in WMMA kernels (128 threads)

__device__ __forceinline__ unsigned short f2bfu(float f) {
  unsigned u = __builtin_bit_cast(unsigned, f);
  return (unsigned short)((u + 0x7FFFu + ((u >> 16) & 1u)) >> 16);   // RNE
}
__device__ __forceinline__ __bf16 f2bf(float f) {
  unsigned short h = f2bfu(f);
  return __builtin_bit_cast(__bf16, h);
}
__device__ __forceinline__ float bf2f(unsigned short h) {
  unsigned u = ((unsigned)h) << 16;
  return __builtin_bit_cast(float, u);
}
__device__ __forceinline__ unsigned pk(unsigned short l, unsigned short h) {
  return (unsigned)l | ((unsigned)h << 16);
}
// branchless sigmoid: v_exp_f32 + v_add + v_rcp_f32
__device__ __forceinline__ float sigmoid_f(float x) {
  return __builtin_amdgcn_rcpf(1.f + __expf(-x));
}
__device__ __forceinline__ float gelu_f(float x) {
  return x * sigmoid_f(1.595769122f * (x + 0.044715f * x * x * x));
}
__device__ __forceinline__ float silu_f(float x) { return x * sigmoid_f(x); }

// ---------------------------------------------------------------------------
// Flat f32 -> bf16 convert (4 elems/thread), n % 4 == 0
// ---------------------------------------------------------------------------
__global__ void f32_to_bf16(const float* __restrict__ s, unsigned short* __restrict__ d,
                            long n) {
  long i4 = ((long)blockIdx.x * blockDim.x + threadIdx.x) * 4;
  if (i4 >= n) return;
  float4 v = *(const float4*)(s + i4);
  uint2 o;
  o.x = pk(f2bfu(v.x), f2bfu(v.y));
  o.y = pk(f2bfu(v.z), f2bfu(v.w));
  *(uint2*)(d + i4) = o;
}

// ---------------------------------------------------------------------------
// Conv weight repack: (CoReal,CiReal,3,3) f32 -> (CoPad,Kpad) bf16,
// k = tap*CiPad + ci, zeros for ci>=CiReal, k>=9*CiPad, m>=CoReal.
// ---------------------------------------------------------------------------
__global__ void prep_conv_w(const float* __restrict__ w, unsigned short* __restrict__ wp,
                            int CoReal, int CoPad, int CiReal, int CiPad, int Kpad) {
  int i = blockIdx.x * blockDim.x + threadIdx.x;
  if (i >= CoPad * Kpad) return;
  int m = i / Kpad, k = i - m * Kpad;
  int tap = k / CiPad, ci = k - tap * CiPad;
  bool ok = (m < CoReal) && (tap < 9) && (ci < CiReal);
  int src = ok ? (m * CiReal + ci) * 9 + tap : 0;
  wp[i] = f2bfu(ok ? w[src] : 0.f);
}

// Bias pad: dst[0..Cop) = i<Co ? b[i] : 0
__global__ void pad_bias(const float* __restrict__ b, float* __restrict__ d,
                         int Co, int Cop) {
  int i = blockIdx.x * blockDim.x + threadIdx.x;
  if (i >= Cop) return;
  d[i] = (i < Co) ? b[i] : 0.f;
}

// ---------------------------------------------------------------------------
// conv0: Ci=1, Co=8 (padded to 16 NHWC channels), direct VALU (75M MACs).
// Patch loads are clamped-address + select (branchless).
// ---------------------------------------------------------------------------
__global__ void conv0_direct(const float* __restrict__ x, const float* __restrict__ w,
                             const float* __restrict__ b, unsigned short* __restrict__ out,
                             int NI, int H, int W) {
  long Np = (long)NI * H * W;
  long i = (long)blockIdx.x * blockDim.x + threadIdx.x;
  if (i >= Np) return;
  long hw = (long)H * W;
  int img = (int)(i / hw); int r = (int)(i - (long)img * hw);
  int oy = r / W, ox = r - oy * W;
  float p[9];
#pragma unroll
  for (int t = 0; t < 9; t++) {
    int ky = t / 3, kx = t - ky * 3, iy = oy + ky - 1, ix = ox + kx - 1;
    bool ok = iy >= 0 && iy < H && ix >= 0 && ix < W;
    int iyc = ok ? iy : 0, ixc = ok ? ix : 0;
    float v = x[(long)img * hw + (long)iyc * W + ixc];
    p[t] = ok ? v : 0.f;
  }
  unsigned short o[16];
#pragma unroll
  for (int c = 0; c < 8; c++) {
    float s = b[c];
#pragma unroll
    for (int t = 0; t < 9; t++) s += w[c * 9 + t] * p[t];
    o[c] = f2bfu(gelu_f(s));
  }
#pragma unroll
  for (int c = 8; c < 16; c++) o[c] = 0;
  uint4* d = (uint4*)(out + i * 16);
  uint4 v0, v1;
  v0.x = pk(o[0], o[1]);  v0.y = pk(o[2], o[3]);  v0.z = pk(o[4], o[5]);  v0.w = pk(o[6], o[7]);
  v1.x = pk(o[8], o[9]);  v1.y = pk(o[10], o[11]); v1.z = pk(o[12], o[13]); v1.w = pk(o[14], o[15]);
  d[0] = v0; d[1] = v1;
}

// ---------------------------------------------------------------------------
// Implicit-GEMM 3x3 conv (stride 1, pad 1) on NHWC bf16, WMMA bf16.
// Weights AND bias zero-padded in output channels -> branchless epilogue.
// ---------------------------------------------------------------------------
__global__ void conv3x3_nhwc_wmma(const unsigned short* __restrict__ x,
                                  const unsigned short* __restrict__ wp,
                                  const float* __restrict__ biap,   // Cop floats, padded
                                  unsigned short* __restrict__ out_nhwc,
                                  int NI, int Ci, int H, int W,
                                  int Kpad, int logCi, int Cop) {
  int lane = threadIdx.x & 31, half = lane >> 4, ln = lane & 15;
  long hw = (long)H * W;
  long Npix = (long)NI * hw;
  long pt = (long)blockIdx.x * WPB + (threadIdx.x >> 5);
  if (pt * 16 >= Npix) return;                       // wave-uniform
  int octile = blockIdx.y;
  long n = pt * 16 + ln;
  int img = (int)(n / hw); int r = (int)(n - (long)img * hw);
  int oy = r / W, ox = r - oy * W;
  int K = Ci * 9;
  const unsigned short* wrow = wp + (long)(octile * 16 + ln) * Kpad;
  v8f acc = {};
  const uint4 z4 = {0u, 0u, 0u, 0u};
  for (int k0 = 0; k0 < K; k0 += 32) {
    uint4 a0 = *(const uint4*)(wrow + k0 + 16 * half);
    uint4 a1 = *(const uint4*)(wrow + k0 + 16 * half + 8);
    U4x2 au = {a0, a1};
    v16bf a = __builtin_bit_cast(v16bf, au);
    int kb0 = k0 + half * 16;
    int tap = kb0 >> logCi;
    int ci0 = kb0 - (tap << logCi);
    int ky = tap / 3, kx = tap - ky * 3;
    int iy = oy + ky - 1, ix = ox + kx - 1;
    bool inb = (tap < 9) && (iy >= 0) && (iy < H) && (ix >= 0) && (ix < W);
    int iyc = inb ? iy : 0, ixc = inb ? ix : 0;      // clamped safe address
    long ep = (((long)img * H + iyc) * W + ixc) * Ci + ci0;
    const uint4* bp = (const uint4*)(x + ep);
    uint4 b0 = bp[0], b1 = bp[1];
    if (!inb) { b0 = z4; b1 = z4; }
    U4x2 bu = {b0, b1};
    v16bf bf = __builtin_bit_cast(v16bf, bu);
    acc = __builtin_amdgcn_wmma_f32_16x16x32_bf16(false, a, false, bf,
                                                  (short)0, acc, false, false);
  }
  int mbase = octile * 16 + half * 8;                // contiguous 8 channels
  float4 bb0 = *(const float4*)(biap + mbase);
  float4 bb1 = *(const float4*)(biap + mbase + 4);
  float rr[8] = {acc[0] + bb0.x, acc[1] + bb0.y, acc[2] + bb0.z, acc[3] + bb0.w,
                 acc[4] + bb1.x, acc[5] + bb1.y, acc[6] + bb1.z, acc[7] + bb1.w};
  unsigned short hs[8];
#pragma unroll
  for (int j = 0; j < 8; j++) hs[j] = f2bfu(gelu_f(rr[j]));  // gelu(0)=0 on pads
  uint4 v;
  v.x = pk(hs[0], hs[1]); v.y = pk(hs[2], hs[3]);
  v.z = pk(hs[4], hs[5]); v.w = pk(hs[6], hs[7]);
  *(uint4*)(out_nhwc + n * Cop + mbase) = v;
}

// ---------------------------------------------------------------------------
// NHWC bf16 (Cp padded channels) -> NCHW f32 repack (conv4 flatten input)
// ---------------------------------------------------------------------------
__global__ void nhwc_to_nchw_f32(const unsigned short* __restrict__ in,
                                 float* __restrict__ out,
                                 int NI, int C, int HW, int Cp) {
  long total = (long)NI * C * HW;
  long i = (long)blockIdx.x * blockDim.x + threadIdx.x;
  if (i >= total) return;
  int p = (int)(i % HW); long t = i / HW;
  int c = (int)(t % C); int img = (int)(t / C);
  out[i] = bf2f(in[((long)img * HW + p) * Cp + c]);
}

// ---------------------------------------------------------------------------
// 2x2 max pool on NHWC bf16 (channels innermost -> coalesced)
// ---------------------------------------------------------------------------
__global__ void maxpool_nhwc(const unsigned short* __restrict__ in,
                             unsigned short* __restrict__ out,
                             int NI, int H, int W, int C) {
  int Ho = H >> 1, Wo = W >> 1;
  long total = (long)NI * Ho * Wo * C;
  long i = (long)blockIdx.x * blockDim.x + threadIdx.x;
  if (i >= total) return;
  int c = (int)(i % C); long t = i / C;
  int xo = (int)(t % Wo); t /= Wo;
  int yo = (int)(t % Ho); int img = (int)(t / Ho);
  const unsigned short* p = in + (((long)img * H + yo * 2) * W + xo * 2) * C + c;
  long rs = (long)W * C;
  float m = fmaxf(fmaxf(bf2f(p[0]), bf2f(p[C])), fmaxf(bf2f(p[rs]), bf2f(p[rs + C])));
  out[i] = f2bfu(m);
}

// ---------------------------------------------------------------------------
// Build 66112-float MLP input: enc(65536, NCHW f32) ++ state(512) ++ action(64)
// ---------------------------------------------------------------------------
__global__ void build_vec(const float* __restrict__ enc, const float* __restrict__ st,
                          const float* __restrict__ act, float* __restrict__ out) {
  int i = blockIdx.x * blockDim.x + threadIdx.x;
  if (i >= 66112) return;
  out[i] = (i < 65536) ? enc[i] : (i < 66048 ? st[i - 65536] : act[i - 66048]);
}

// ---------------------------------------------------------------------------
// Memory-bound GEMV (float4 streams): out[row] = act(W[row,:]·v + b[row])
// ---------------------------------------------------------------------------
__global__ void gemv_act(const float* __restrict__ W, const float* __restrict__ v,
                         const float* __restrict__ b, float* __restrict__ out,
                         int K, int act) {
  __shared__ float red[256];
  int row = blockIdx.x;
  const float4* Wr = (const float4*)(W + (long)row * K);
  const float4* vv = (const float4*)v;
  int K4 = K >> 2;
  float s = 0.f;
  for (int k = threadIdx.x; k < K4; k += 256) {
    float4 a = Wr[k], x = vv[k];
    s += a.x * x.x + a.y * x.y + a.z * x.z + a.w * x.w;
  }
  red[threadIdx.x] = s; __syncthreads();
  for (int o = 128; o > 0; o >>= 1) {
    if (threadIdx.x < o) red[threadIdx.x] += red[threadIdx.x + o];
    __syncthreads();
  }
  if (threadIdx.x == 0) {
    float r = red[0] + (b ? b[row] : 0.f);
    out[row] = (act == 1) ? gelu_f(r) : r;
  }
}

// ---------------------------------------------------------------------------
// out[16,N] = act(Xb[16,K]·Wb[N,K]^T + bias), both operands bf16 (pre-staged).
// N%16==0, K%32==0. Loop: 4x b128 + 1 wmma. Epilogue: float4 bias + stores.
// ---------------------------------------------------------------------------
__global__ void gemm_bf16_wmma(const unsigned short* __restrict__ Xb,
                               const unsigned short* __restrict__ Wb,
                               const float* __restrict__ bias, float* __restrict__ out,
                               int N, int K, int act) {
  int lane = threadIdx.x & 31, half = lane >> 4, ln = lane & 15;
  int ntile = blockIdx.x * WPB + (threadIdx.x >> 5);
  if (ntile * 16 >= N) return;                       // wave-uniform
  const unsigned short* wrow = Wb + (long)(ntile * 16 + ln) * K;
  const unsigned short* xrow = Xb + (long)ln * K;
  v8f acc = {};
  for (int k0 = 0; k0 < K; k0 += 32) {
    uint4 a0 = *(const uint4*)(wrow + k0 + 16 * half);
    uint4 a1 = *(const uint4*)(wrow + k0 + 16 * half + 8);
    U4x2 au = {a0, a1};
    v16bf a = __builtin_bit_cast(v16bf, au);
    uint4 b0 = *(const uint4*)(xrow + k0 + 16 * half);
    uint4 b1 = *(const uint4*)(xrow + k0 + 16 * half + 8);
    U4x2 bu = {b0, b1};
    v16bf bf = __builtin_bit_cast(v16bf, bu);
    acc = __builtin_amdgcn_wmma_f32_16x16x32_bf16(false, a, false, bf,
                                                  (short)0, acc, false, false);
  }
  int mbase = ntile * 16 + half * 8;                 // contiguous 8 outputs
  float4 bb0 = {0.f, 0.f, 0.f, 0.f}, bb1 = {0.f, 0.f, 0.f, 0.f};
  if (bias) {
    bb0 = *(const float4*)(bias + mbase);
    bb1 = *(const float4*)(bias + mbase + 4);
  }
  float rr[8] = {acc[0] + bb0.x, acc[1] + bb0.y, acc[2] + bb0.z, acc[3] + bb0.w,
                 acc[4] + bb1.x, acc[5] + bb1.y, acc[6] + bb1.z, acc[7] + bb1.w};
  if (act == 1) {
#pragma unroll
    for (int j = 0; j < 8; j++) rr[j] = gelu_f(rr[j]);
  }
  float4 s0 = {rr[0], rr[1], rr[2], rr[3]};
  float4 s1 = {rr[4], rr[5], rr[6], rr[7]};
  float* orow = out + (long)ln * N + mbase;          // transposed store
  *(float4*)(orow) = s0;
  *(float4*)(orow + 4) = s1;
}

// --------------------------- Mamba2 pieces ---------------------------------
__global__ void dt_softplus(const float* __restrict__ P, const float* __restrict__ dtbias,
                            float* __restrict__ dt) {
  int i = threadIdx.x + blockIdx.x * blockDim.x;
  if (i >= 512) return;
  int t = i >> 5, h = i & 31;
  float x = P[t * 4384 + 4352 + h] + dtbias[h];
  dt[i] = (x > 20.f) ? x : log1pf(expf(x));
}

__global__ void conv1d_silu(const float* __restrict__ P, const float* __restrict__ cw,
                            const float* __restrict__ cb, float* __restrict__ xbcc) {
  int i = threadIdx.x + blockIdx.x * blockDim.x;
  if (i >= 16 * 2304) return;
  int c = i % 2304, t = i / 2304;
  float s = cb[c];
#pragma unroll
  for (int k = 0; k < 4; k++) {
    int tt = t - 3 + k;
    if (tt >= 0) s += P[tt * 4384 + 2048 + c] * cw[c * 4 + k];
  }
  xbcc[i] = silu_f(s);
}

// Closed-form SSM scan: dA = exp(dt*A) exactly, so
// y[t,p] = sum_{s<=t} exp(A*(cumdt[t]-cumdt[s]))*dt[s]*(C_t·B_s)*x[s,p]
// -> M (16x16) built in LDS, K padded 16->32, one WMMA per 16-wide p tile.
__global__ void ssm_head_wmma(const float* __restrict__ xbcc, const float* __restrict__ dt,
                              const float* __restrict__ A_log, const float* __restrict__ Dp,
                              float* __restrict__ y) {
  __shared__ float Mm[16][16];
  __shared__ float cumL[16], dts[16];
  int h = blockIdx.x, lane = threadIdx.x;
  int half = lane >> 4, ln = lane & 15;
  float Ah = -expf(A_log[h]);
  float dl = (lane < 16) ? dt[lane * 32 + h] : 0.f;
  float c = dl;
  for (int d = 1; d < 16; d <<= 1) {                 // inclusive prefix over t
    float o = __shfl_up(c, d, 32);
    if (lane >= d) c += o;
  }
  if (lane < 16) { cumL[lane] = c; dts[lane] = dl; }
  __syncthreads();
  for (int q = 0; q < 8; q++) {
    int p = lane * 8 + q; int t = p >> 4, s = p & 15;
    float g = 0.f;
    if (s <= t) {
      const float* Bv = xbcc + s * 2304 + 2048;
      const float* Cv = xbcc + t * 2304 + 2176;
      for (int n = 0; n < 128; n++) g += Bv[n] * Cv[n];
      g *= __expf(Ah * (cumL[t] - cumL[s])) * dts[s];
    }
    Mm[t][s] = g;
  }
  __syncthreads();
  v16bf a;
#pragma unroll
  for (int i = 0; i < 16; i++) {
    int k = i + half * 16;
    a[i] = f2bf((k < 16) ? Mm[ln][k] : 0.f);
  }
  float Dh = Dp[h];
  for (int pt = 0; pt < 4; pt++) {
    v16bf bf;
    int p = pt * 16 + ln;
#pragma unroll
    for (int i = 0; i < 16; i++) {
      int k = i + half * 16;
      bf[i] = f2bf((k < 16) ? xbcc[k * 2304 + h * 64 + p] : 0.f);
    }
    v8f acc = {};
    acc = __builtin_amdgcn_wmma_f32_16x16x32_bf16(false, a, false, bf,
                                                  (short)0, acc, false, false);
#pragma unroll
    for (int j = 0; j < 8; j++) {
      int t = j + half * 8;
      float xv = xbcc[t * 2304 + h * 64 + p];
      y[t * 2048 + h * 64 + p] = acc[j] + Dh * xv;
    }
  }
}

__global__ void gate_rmsnorm(float* __restrict__ y, const float* __restrict__ P,
                             const float* __restrict__ nw) {
  __shared__ float red[256];
  int t = blockIdx.x;
  float g[8]; float ss = 0.f;
  for (int q = 0; q < 8; q++) {
    int c = threadIdx.x + q * 256;
    float v = y[t * 2048 + c] * silu_f(P[t * 4384 + c]);
    g[q] = v; ss += v * v;
  }
  red[threadIdx.x] = ss; __syncthreads();
  for (int o = 128; o > 0; o >>= 1) {
    if (threadIdx.x < o) red[threadIdx.x] += red[threadIdx.x + o];
    __syncthreads();
  }
  float rstd = rsqrtf(red[0] / 2048.f + 1e-5f);
  for (int q = 0; q < 8; q++) {
    int c = threadIdx.x + q * 256;
    y[t * 2048 + c] = g[q] * rstd * nw[c];
  }
}

// ---------------------------------------------------------------------------
// ConvTranspose2d(k=4,s=2,p=1)+GELU. Parity: ky ≡ oy (mod 2) -> 2x2 taps.
// ---------------------------------------------------------------------------
__global__ void deconv4x4_gelu(const float* __restrict__ x, const float* __restrict__ w,
                               const float* __restrict__ b, float* __restrict__ out,
                               int NI, int Ci, int Co, int H, int W) {
  int Ho = 2 * H, Wo = 2 * W;
  long total = (long)NI * Co * Ho * Wo;
  long i = (long)blockIdx.x * blockDim.x + threadIdx.x;
  if (i >= total) return;
  int ox = (int)(i % Wo); long r = i / Wo;
  int oy = (int)(r % Ho); r /= Ho;
  int co = (int)(r % Co); int f = (int)(r / Co);
  int py = oy & 1, px = ox & 1;
  int iy0 = (oy + py - 2) >> 1, ix0 = (ox + px - 2) >> 1;
  long hw = (long)H * W;
  float acc = b[co];
#pragma unroll
  for (int a = 0; a < 2; a++) {
    int iy = iy0 + a, ky = py + 2 * a;
    if (iy < 0 || iy >= H) continue;
#pragma unroll
    for (int bb = 0; bb < 2; bb++) {
      int ix = ix0 + bb, kx = px + 2 * bb;
      if (ix < 0 || ix >= W) continue;
      const float* xp = x + (long)f * Ci * hw + (long)iy * W + ix;
      const float* wt = w + (long)co * Ci * 16 + ky * 4 + kx;
      float s = 0.f;
      for (int ci = 0; ci < Ci; ci++) s += xp[ci * hw] * wt[ci * 16];
      acc += s;
    }
  }
  out[i] = gelu_f(acc);
}

// ---------------------------------------------------------------------------
extern "C" void kernel_launch(void* const* d_in, const int* in_sizes, int n_in,
                              void* d_out, int out_size, void* d_ws, size_t ws_size,
                              hipStream_t stream) {
  // Flattening assumed in setup_inputs() insertion order:
  // 0 obs, 1 action, 2 state, 3..12 enc (w,b)x5, 13..20 lr_mlp (w,b)x4,
  // 21..22 state_enc, 23+8L mamba {in_proj,conv_w,conv_b,dt_bias,A_log,D,
  // norm_w,out_proj}, 55..58 dec_mlp, 59..64 dcnn.
  const float* obs = (const float*)d_in[0];
  const float* act = (const float*)d_in[1];
  const float* st  = (const float*)d_in[2];
  auto Wp = [&](int i) { return (const float*)d_in[i]; };

  // ---- workspace layout (bump allocator, 256B aligned) ----
  char* base = (char*)d_ws;
  size_t off = 0;
  auto alloc = [&](size_t bytes) { void* p = base + off; off += (bytes + 255) & ~size_t(255); return p; };
  unsigned short* actA = (unsigned short*)alloc(2UL * 16 * 256 * 256 * 64); // 134MB
  unsigned short* actB = (unsigned short*)alloc(2UL * 16 * 256 * 256 * 64); // 134MB
  unsigned short* wp1 = (unsigned short*)alloc(2UL * 64 * 160);
  unsigned short* wp2 = (unsigned short*)alloc(2UL * 128 * 576);
  unsigned short* wp3 = (unsigned short*)alloc(2UL * 32 * 1152);
  unsigned short* wp4 = (unsigned short*)alloc(2UL * 16 * 288);
  float* bp1 = (float*)alloc(4UL * 64);
  float* bp2 = (float*)alloc(4UL * 128);
  float* bp3 = (float*)alloc(4UL * 32);
  float* bp4 = (float*)alloc(4UL * 16);
  unsigned short* wbIn  = (unsigned short*)alloc(2UL * 4384 * 1024); // in_proj bf16
  unsigned short* wbOut = (unsigned short*)alloc(2UL * 1024 * 2048); // out_proj bf16
  unsigned short* wbD1  = (unsigned short*)alloc(2UL * 1024 * 1024);
  unsigned short* wbD2  = (unsigned short*)alloc(2UL * 1024 * 1024);
  unsigned short* ztb   = (unsigned short*)alloc(2UL * 16384);
  unsigned short* tmpb  = (unsigned short*)alloc(2UL * 16384);
  unsigned short* ybufb = (unsigned short*)alloc(2UL * 32768);
  float* c4out = (float*)alloc(4UL * 65536);           // conv4 NCHW f32
  float* vec   = (float*)alloc(4UL * 66112);
  float* va    = (float*)alloc(4UL * 1024);
  float* vb    = (float*)alloc(4UL * 1024);
  float* zt    = (float*)alloc(4UL * 16384);
  float* tmp   = (float*)alloc(4UL * 16384);
  float* P     = (float*)alloc(4UL * 70144);
  float* xbcc  = (float*)alloc(4UL * 36864);
  float* dtb   = (float*)alloc(4UL * 512);
  float* ybuf  = (float*)alloc(4UL * 32768);
  float* dd1   = (float*)alloc(4UL * 2097152);
  float* dd2   = (float*)alloc(4UL * 2097152);
  float* img    = (float*)d_out;
  float* stateo = img + 1048576;

  dim3 blk(128);
  auto cg = [&](long npix, int ctiles) {
    return dim3((unsigned)((npix + 16L * WPB - 1) / (16L * WPB)), (unsigned)ctiles);
  };
  auto cvt = [&](const float* s, unsigned short* d, long n) {
    f32_to_bf16<<<(unsigned)((n / 4 + 255) / 256), 256, 0, stream>>>(s, d, n);
  };

  // ---- weight/bias staging (padded rows + padded bias -> branchless conv) ----
  prep_conv_w<<<(64 * 160 + 255) / 256, 256, 0, stream>>>(Wp(5),  wp1, 64, 64, 8, 16, 160);
  prep_conv_w<<<(128 * 576 + 255) / 256, 256, 0, stream>>>(Wp(7),  wp2, 128, 128, 64, 64, 576);
  prep_conv_w<<<(32 * 1152 + 255) / 256, 256, 0, stream>>>(Wp(9),  wp3, 32, 32, 128, 128, 1152);
  prep_conv_w<<<(16 * 288 + 255) / 256, 256, 0, stream>>>(Wp(11), wp4, 4, 16, 32, 32, 288);
  pad_bias<<<1, 64,  0, stream>>>(Wp(6),  bp1, 64, 64);
  pad_bias<<<1, 128, 0, stream>>>(Wp(8),  bp2, 128, 128);
  pad_bias<<<1, 32,  0, stream>>>(Wp(10), bp3, 32, 32);
  pad_bias<<<1, 16,  0, stream>>>(Wp(12), bp4, 4, 16);
  cvt(Wp(55), wbD1, 1024L * 1024);
  cvt(Wp(57), wbD2, 1024L * 1024);

  // ---- encoder ----
  conv0_direct<<<(16L * 256 * 256 + 255) / 256, 256, 0, stream>>>(obs, Wp(3), Wp(4), actA, 16, 256, 256);
  conv3x3_nhwc_wmma<<<cg(16L * 256 * 256, 4), blk, 0, stream>>>(actA, wp1, bp1, actB,
                                                                16, 16, 256, 256, 160, 4, 64);
  { long t = 16L * 128 * 128 * 64;  maxpool_nhwc<<<(t + 255) / 256, 256, 0, stream>>>(actB, actA, 16, 256, 256, 64); }
  conv3x3_nhwc_wmma<<<cg(16L * 128 * 128, 8), blk, 0, stream>>>(actA, wp2, bp2, actB,
                                                                16, 64, 128, 128, 576, 6, 128);
  { long t = 16L * 64 * 64 * 128;   maxpool_nhwc<<<(t + 255) / 256, 256, 0, stream>>>(actB, actA, 16, 128, 128, 128); }
  conv3x3_nhwc_wmma<<<cg(16L * 64 * 64, 2), blk, 0, stream>>>(actA, wp3, bp3, actB,
                                                              16, 128, 64, 64, 1152, 7, 32);
  { long t = 16L * 32 * 32 * 32;    maxpool_nhwc<<<(t + 255) / 256, 256, 0, stream>>>(actB, actA, 16, 64, 64, 32); }
  conv3x3_nhwc_wmma<<<cg(16L * 32 * 32, 1), blk, 0, stream>>>(actA, wp4, bp4, actB,
                                                              16, 32, 32, 32, 288, 5, 16);
  nhwc_to_nchw_f32<<<(65536 + 255) / 256, 256, 0, stream>>>(actB, c4out, 16, 4, 1024, 16);

  // ---- batch-1 latent MLP (memory-bound GEMV) ----
  build_vec<<<(66112 + 255) / 256, 256, 0, stream>>>(c4out, st, act, vec);
  gemv_act<<< 1024, 256, 0, stream>>>(Wp(13), vec, Wp(14), va, 66112, 1);
  gemv_act<<< 1024, 256, 0, stream>>>(Wp(15), va,  Wp(16), vb,  1024, 1);
  gemv_act<<< 1024, 256, 0, stream>>>(Wp(17), vb,  Wp(18), va,  1024, 1);
  gemv_act<<<16384, 256, 0, stream>>>(Wp(19), va,  Wp(20), zt,  1024, 0);  // -> (16,1024)

  // ---- 4 Mamba2 mixers ----
  for (int L = 0; L < 4; L++) {
    int b0 = 23 + 8 * L;
    cvt(Wp(b0 + 0), wbIn, 4384L * 1024);
    cvt(zt, ztb, 16384);
    gemm_bf16_wmma<<<(4384 / 16 + WPB - 1) / WPB, blk, 0, stream>>>(ztb, wbIn, nullptr, P, 4384, 1024, 0);
    dt_softplus<<<2, 256, 0, stream>>>(P, Wp(b0 + 3), dtb);
    conv1d_silu<<<(16 * 2304 + 255) / 256, 256, 0, stream>>>(P, Wp(b0 + 1), Wp(b0 + 2), xbcc);
    ssm_head_wmma<<<32, 32, 0, stream>>>(xbcc, dtb, Wp(b0 + 4), Wp(b0 + 5), ybuf);
    gate_rmsnorm<<<16, 256, 0, stream>>>(ybuf, P, Wp(b0 + 6));
    cvt(Wp(b0 + 7), wbOut, 1024L * 2048);
    cvt(ybuf, ybufb, 32768);
    gemm_bf16_wmma<<<(1024 / 16 + WPB - 1) / WPB, blk, 0, stream>>>(ybufb, wbOut, nullptr, zt, 1024, 2048, 0);
  }

  // ---- state head ----
  gemv_act<<<512, 256, 0, stream>>>(Wp(21), zt + 15 * 1024, Wp(22), stateo, 1024, 0);

  // ---- decoder MLP + deconvs ----
  cvt(zt, ztb, 16384);
  gemm_bf16_wmma<<<(1024 / 16 + WPB - 1) / WPB, blk, 0, stream>>>(ztb, wbD1, Wp(56), tmp, 1024, 1024, 1);
  cvt(tmp, tmpb, 16384);
  gemm_bf16_wmma<<<(1024 / 16 + WPB - 1) / WPB, blk, 0, stream>>>(tmpb, wbD2, Wp(58), P, 1024, 1024, 0);
  { long t = 16L * 32 * 64 * 64;  deconv4x4_gelu<<<(t + 255) / 256, 256, 0, stream>>>(P,   Wp(59), Wp(60), dd1, 16, 1, 32, 32, 32); }
  { long t = 16L * 8 * 128 * 128; deconv4x4_gelu<<<(t + 255) / 256, 256, 0, stream>>>(dd1, Wp(61), Wp(62), dd2, 16, 32, 8, 64, 64); }
  { long t = 16L * 1 * 256 * 256; deconv4x4_gelu<<<(t + 255) / 256, 256, 0, stream>>>(dd2, Wp(63), Wp(64), img, 16, 8, 1, 128, 128); }
}
